// LightGCL_51668456571541
// MI455X (gfx1250) — compile-verified, matrix operand order checked by reading, not run on hardware
//
#include <hip/hip_runtime.h>
#include <hip/hip_bf16.h>
#include <stdint.h>

// ---------------- problem constants (match reference) ----------------
#define NU     100000
#define NI     50000
#define DIM    128
#define LAYERS 2
#define QQ     5
#define NEDGE  1000000
#define BB     4096

__device__ __host__ constexpr float kTEMP    = 0.2f;
__device__ __host__ constexpr float kLAMBDA1 = 0.2f;
__device__ __host__ constexpr float kKEEP    = 0.8f;   // 1 - dropout

// ---------------- CDNA5 WMMA / TDM types ----------------
typedef __attribute__((ext_vector_type(16))) _Float16 v16h;
typedef __attribute__((ext_vector_type(8)))  float    v8f;
typedef __attribute__((ext_vector_type(8)))  _Float16 half8v;
typedef __attribute__((ext_vector_type(4)))  int      i32x4;
typedef __attribute__((ext_vector_type(8)))  int      i32x8;

// ---------------- small device helpers ----------------
__device__ inline float lrelu(float x) { return x > 0.f ? x : 0.5f * x; }

__device__ inline float u01(uint32_t x) {
    x ^= x >> 16; x *= 0x7feb352dU;
    x ^= x >> 15; x *= 0x846ca68bU;
    x ^= x >> 16;
    return (float)(x >> 8) * (1.0f / 16777216.0f);
}

__device__ inline void atomAddF(float* p, float v) {
    __hip_atomic_fetch_add(p, v, __ATOMIC_RELAXED, __HIP_MEMORY_SCOPE_AGENT);
}

// Load one wave's slice of a 16x32 f16 A/B operand for v_wmma_f32_16x16x32_f16.
// Row-major source, row = (tile*16 + (lane&15)), k window [k0, k0+32).
// Lane layout (ISA 7.12.2): lanes 0-15 hold K = {k0..k0+7, k0+16..k0+23},
// lanes 16-31 hold K = {k0+8..k0+15, k0+24..k0+31}. Two 16-byte loads per lane.
__device__ inline v16h load_tile16(const _Float16* __restrict__ rowptr, int k0, int lane) {
    const int kb = k0 + ((lane >> 4) << 3);
    half8v lo = *reinterpret_cast<const half8v*>(rowptr + kb);
    half8v hi = *reinterpret_cast<const half8v*>(rowptr + kb + 16);
    v16h r;
#pragma unroll
    for (int j = 0; j < 8; ++j) { r[j] = lo[j]; r[j + 8] = hi[j]; }
    return r;
}

// Issue a TDM load of a 2-D f16 tile (rows x DIM, row-major, stride DIM) from
// global memory into LDS. D# per ISA 8.3/8.4: group0 = {count, lds_addr,
// global_addr, type=2}; group1 = {data_size=2B, tensor dims/strides, tile dims}.
// Groups 2/3 unused for 2-D tensors (pass a zero quad for both).
__device__ inline void tdm_load_2d_f16(const _Float16* gsrc, uint32_t lds_addr,
                                       int rows) {
    const uint64_t ga = (uint64_t)(uintptr_t)gsrc;
    i32x4 g0;
    g0[0] = 1;                                    // count=1 valid descriptor
    g0[1] = (int)lds_addr;                        // lds_addr[31:0]
    g0[2] = (int)(uint32_t)(ga & 0xffffffffu);    // global_addr[31:0]
    g0[3] = (int)((uint32_t)((ga >> 32) & 0x01ffffffu) | 0x80000000u); // addr[56:32] | type=2
    i32x8 g1;
    g1[0] = 0x00010000;                           // workgroup_mask=0, data_size=1 (2B)
    g1[1] = (int)(128u << 16);                    // tensor_dim0 = 128 (bits 79:48 lo)
    g1[2] = (int)(4096u << 16);                   // tensor_dim1 = 4096 (bits 111:80 lo)
    g1[3] = (int)(128u << 16);                    // tile_dim0 = 128 (bits 127:112)
    g1[4] = rows;                                 // tile_dim1 (bits 143:128)
    g1[5] = 128;                                  // tensor_dim0_stride = 128 (bits 207:160 lo)
    g1[6] = 0;
    g1[7] = 0;
    i32x4 gz;
    gz[0] = 0; gz[1] = 0; gz[2] = 0; gz[3] = 0;
    asm volatile("tensor_load_to_lds %0, %1, %2, %3"
                 :: "s"(g0), "s"(g1), "s"(gz), "s"(gz)
                 : "memory");
}

// ---------------- kernels ----------------

// Z[dst[e]] += mask(e) * vals[e]/keep * Esrc[src[e]]  (one wave per edge)
__global__ void spmm_scatter(const float* __restrict__ vals,
                             const int*  __restrict__ dst,
                             const int*  __restrict__ src,
                             const float* __restrict__ Esrc,
                             float* __restrict__ Z,
                             int nE, uint32_t seed) {
    const int lane   = threadIdx.x & 31;
    const int wave   = (blockIdx.x * blockDim.x + threadIdx.x) >> 5;
    const int nwaves = (gridDim.x * blockDim.x) >> 5;
    for (int e = wave; e < nE; e += nwaves) {
        if (u01(seed ^ (0x9E3779B9u * (uint32_t)e)) >= kKEEP) continue;  // dropped
        const float v = vals[e] * (1.0f / kKEEP);
        const int r = dst[e], c = src[e];
        float4 x = reinterpret_cast<const float4*>(Esrc + (size_t)c * DIM)[lane];
        float* p = Z + (size_t)r * DIM + lane * 4;
        atomAddF(p + 0, x.x * v);
        atomAddF(p + 1, x.y * v);
        atomAddF(p + 2, x.z * v);
        atomAddF(p + 3, x.w * v);
    }
}

// P[q][d] += sum_n vt[q][n] * E[n][d]   (P is QQ x DIM, pre-zeroed)
__global__ void qd_gemm(const float* __restrict__ vt, const float* __restrict__ E,
                        float* __restrict__ P, int nrows) {
    const int d = threadIdx.x;  // blockDim.x == DIM
    float acc[QQ];
#pragma unroll
    for (int q = 0; q < QQ; ++q) acc[q] = 0.f;
    for (int n = blockIdx.x; n < nrows; n += gridDim.x) {
        const float e = E[(size_t)n * DIM + d];
#pragma unroll
        for (int q = 0; q < QQ; ++q) acc[q] += vt[(size_t)q * nrows + n] * e;
    }
#pragma unroll
    for (int q = 0; q < QQ; ++q) atomAddF(&P[q * DIM + d], acc[q]);
}

// G[n][d] = lrelu( sum_q mul[n][q] * P[q][d] )
__global__ void gprop(const float* __restrict__ mul, const float* __restrict__ P,
                      float* __restrict__ G, int nrows) {
    __shared__ float sP[QQ * DIM];
    for (int i = threadIdx.x; i < QQ * DIM; i += blockDim.x) sP[i] = P[i];
    __syncthreads();
    const int total = nrows * DIM;
    for (int idx = blockIdx.x * blockDim.x + threadIdx.x; idx < total;
         idx += gridDim.x * blockDim.x) {
        const int n = idx >> 7, d = idx & (DIM - 1);
        float a = 0.f;
#pragma unroll
        for (int q = 0; q < QQ; ++q) a += mul[n * QQ + q] * sP[q * DIM + d];
        G[idx] = lrelu(a);
    }
}

// Z = lrelu(Z); E += Z; Esum += E
__global__ void layer_update(float* __restrict__ Z, float* __restrict__ E,
                             float* __restrict__ Esum, int total) {
    for (int idx = blockIdx.x * blockDim.x + threadIdx.x; idx < total;
         idx += gridDim.x * blockDim.x) {
        const float z = lrelu(Z[idx]);
        Z[idx] = z;
        const float e = E[idx] + z;
        E[idx] = e;
        Esum[idx] += e;
    }
}

// out[b] = normalize(X[ids[b]]); writes f32 (optional) and f16 (optional)
__global__ void gather_norm(const float* __restrict__ X, const int* __restrict__ ids,
                            float* __restrict__ outF, _Float16* __restrict__ outH, int nb) {
    const int lane = threadIdx.x & 31;
    const int b    = (blockIdx.x * blockDim.x + threadIdx.x) >> 5;
    if (b >= nb) return;
    const int id = ids[b];
    float4 x = reinterpret_cast<const float4*>(X + (size_t)id * DIM)[lane];
    float ss = x.x * x.x + x.y * x.y + x.z * x.z + x.w * x.w;
#pragma unroll
    for (int m = 16; m >= 1; m >>= 1) ss += __shfl_xor(ss, m, 32);
    const float inv = 1.0f / fmaxf(sqrtf(ss), 1e-12f);
    x.x *= inv; x.y *= inv; x.z *= inv; x.w *= inv;
    if (outF) reinterpret_cast<float4*>(outF + (size_t)b * DIM)[lane] = x;
    if (outH) {
        _Float16* p = outH + (size_t)b * DIM + lane * 4;
        p[0] = (_Float16)x.x; p[1] = (_Float16)x.y;
        p[2] = (_Float16)x.z; p[3] = (_Float16)x.w;
    }
}

// Wt_h[n][k] = (f16) W[k][n]   (W is DIM x DIM row-major)
__global__ void convert_wt(const float* __restrict__ W, _Float16* __restrict__ Wt) {
    const int idx = blockIdx.x * blockDim.x + threadIdx.x;  // n*DIM + k
    if (idx >= DIM * DIM) return;
    const int n = idx >> 7, k = idx & (DIM - 1);
    Wt[idx] = (_Float16)W[k * DIM + n];
}

// C[m][n] = sum_k A[m][k] * Wt[n][k]   (BB x DIM) = (BB x DIM) @ (DIM x DIM)
// one wave per 16x16 tile; WMMA f16->f32
__global__ void hyper_gemm(const _Float16* __restrict__ A, const _Float16* __restrict__ Bt,
                           float* __restrict__ Cf, _Float16* __restrict__ Ch) {
    const int lane = threadIdx.x & 31;
    const int tm = blockIdx.x, tn = blockIdx.y;
    const _Float16* arow = A  + (size_t)(tm * 16 + (lane & 15)) * DIM;
    const _Float16* brow = Bt + (size_t)(tn * 16 + (lane & 15)) * DIM;
    v8f c = {};
#pragma unroll
    for (int k0 = 0; k0 < DIM; k0 += 32) {
        v16h a = load_tile16(arow, k0, lane);
        v16h b = load_tile16(brow, k0, lane);
        c = __builtin_amdgcn_wmma_f32_16x16x32_f16(false, a, false, b, (short)0, c,
                                                   false, false);
    }
    const int n     = tn * 16 + (lane & 15);
    const int mbase = tm * 16 + ((lane >> 4) << 3);
#pragma unroll
    for (int r = 0; r < 8; ++r) {
        const float v = c[r];
        Cf[(size_t)(mbase + r) * DIM + n] = v;
        Ch[(size_t)(mbase + r) * DIM + n] = (_Float16)v;
    }
}

// pos[b] = exp( dot(gnn[b], hyp[b]) / T )
__global__ void pos_kernel(const float* __restrict__ gnn, const float* __restrict__ hyp,
                           float* __restrict__ pos, int nb) {
    const int lane = threadIdx.x & 31;
    const int b    = (blockIdx.x * blockDim.x + threadIdx.x) >> 5;
    if (b >= nb) return;
    float4 g = reinterpret_cast<const float4*>(gnn + (size_t)b * DIM)[lane];
    float4 h = reinterpret_cast<const float4*>(hyp + (size_t)b * DIM)[lane];
    float d = g.x * h.x + g.y * h.y + g.z * h.z + g.w * h.w;
#pragma unroll
    for (int m = 16; m >= 1; m >>= 1) d += __shfl_xor(d, m, 32);
    if (lane == 0) pos[b] = __expf(d * (1.0f / kTEMP));
}

// neg[i] = sum_j exp( dot(gnn[i], hyp[j]) / T )
// Block = 8 waves; wave w owns row tile (blockIdx.x*8 + w), A tile in registers.
// B panels (64 rows x 128 cols, f16, 16 KB) are streamed into LDS by the TDM
// (tensor_load_to_lds), double-buffered, and consumed by all 8 waves via
// ds_load + v_wmma. exp/row-sum epilogue fused; each wave owns its 16 rows so
// results are plain stores (no atomics).
#define NEG_CHUNK_ROWS 64
#define NEG_NCHUNKS    (BB / NEG_CHUNK_ROWS)
__global__ __launch_bounds__(256) void neg_gemm(const _Float16* __restrict__ A,
                                                const _Float16* __restrict__ Bm,
                                                float* __restrict__ neg) {
    __shared__ _Float16 sB[2][NEG_CHUNK_ROWS * DIM];   // 2 x 16 KB
    const int lane = threadIdx.x & 31;
    const int wv   = threadIdx.x >> 5;                 // 0..7
    const int tm   = blockIdx.x * 8 + wv;              // row tile of this wave

    const _Float16* arow = A + (size_t)(tm * 16 + (lane & 15)) * DIM;
    const v16h a0 = load_tile16(arow,  0, lane);
    const v16h a1 = load_tile16(arow, 32, lane);
    const v16h a2 = load_tile16(arow, 64, lane);
    const v16h a3 = load_tile16(arow, 96, lane);

    float acc[8];
#pragma unroll
    for (int r = 0; r < 8; ++r) acc[r] = 0.f;

    const uint32_t lds0 = (uint32_t)(uintptr_t)&sB[0][0];
    const uint32_t lds1 = (uint32_t)(uintptr_t)&sB[1][0];

    // prologue: fill buffer 0
    if (wv == 0) {
        tdm_load_2d_f16(Bm, lds0, NEG_CHUNK_ROWS);
        __builtin_amdgcn_s_wait_tensorcnt(0);
    }
    __syncthreads();

    for (int ch = 0; ch < NEG_NCHUNKS; ++ch) {
        const int cur = ch & 1;
        // kick off next panel into the other buffer while we compute
        if (wv == 0 && (ch + 1) < NEG_NCHUNKS) {
            tdm_load_2d_f16(Bm + (size_t)(ch + 1) * NEG_CHUNK_ROWS * DIM,
                            cur ? lds0 : lds1, NEG_CHUNK_ROWS);
        }
        const _Float16* panel = &sB[cur][0];
#pragma unroll
        for (int t = 0; t < NEG_CHUNK_ROWS / 16; ++t) {
            const _Float16* brow = panel + (size_t)(t * 16 + (lane & 15)) * DIM;
            v8f c = {};
            c = __builtin_amdgcn_wmma_f32_16x16x32_f16(false, a0, false,
                    load_tile16(brow,  0, lane), (short)0, c, false, false);
            c = __builtin_amdgcn_wmma_f32_16x16x32_f16(false, a1, false,
                    load_tile16(brow, 32, lane), (short)0, c, false, false);
            c = __builtin_amdgcn_wmma_f32_16x16x32_f16(false, a2, false,
                    load_tile16(brow, 64, lane), (short)0, c, false, false);
            c = __builtin_amdgcn_wmma_f32_16x16x32_f16(false, a3, false,
                    load_tile16(brow, 96, lane), (short)0, c, false, false);
#pragma unroll
            for (int r = 0; r < 8; ++r) acc[r] += __expf(c[r] * (1.0f / kTEMP));
        }
        __syncthreads();                               // done reading sB[cur]
        if (wv == 0 && (ch + 1) < NEG_NCHUNKS)
            __builtin_amdgcn_s_wait_tensorcnt(0);      // next panel landed
        __syncthreads();
    }

    // each wave swept all columns: reduce over the 16 lanes sharing rows, store
#pragma unroll
    for (int r = 0; r < 8; ++r) {
        float v = acc[r];
        v += __shfl_xor(v, 1, 32);
        v += __shfl_xor(v, 2, 32);
        v += __shfl_xor(v, 4, 32);
        v += __shfl_xor(v, 8, 32);
        if ((lane & 15) == 0)
            neg[tm * 16 + ((lane >> 4) << 3) + r] = v;
    }
}

// loss_s += sum_b mask(b) * -log(pos/(neg+1e-8) + 1e-8)
__global__ void infonce_loss(const float* __restrict__ pos, const float* __restrict__ neg,
                             float* __restrict__ acc, uint32_t seed, int nb) {
    const int b = blockIdx.x * blockDim.x + threadIdx.x;
    if (b >= nb) return;
    const float mask = (u01(seed ^ (0x85ebca6bu * (uint32_t)b)) > 0.5f) ? 1.f : 0.f;
    const float term = -logf(pos[b] / (neg[b] + 1e-8f) + 1e-8f) * mask;
    atomAddF(acc, term);
}

// loss_r_sum += relu(1 - dot(u, pos) + dot(u, neg))
__global__ void bpr_kernel(const float* __restrict__ Eus, const float* __restrict__ Eis,
                           const int* __restrict__ uids, const int* __restrict__ pids,
                           const int* __restrict__ nids, float* __restrict__ acc, int nb) {
    const int lane = threadIdx.x & 31;
    const int b    = (blockIdx.x * blockDim.x + threadIdx.x) >> 5;
    if (b >= nb) return;
    float4 u = reinterpret_cast<const float4*>(Eus + (size_t)uids[b] * DIM)[lane];
    float4 p = reinterpret_cast<const float4*>(Eis + (size_t)pids[b] * DIM)[lane];
    float4 n = reinterpret_cast<const float4*>(Eis + (size_t)nids[b] * DIM)[lane];
    float ps = u.x * p.x + u.y * p.y + u.z * p.z + u.w * p.w;
    float ns = u.x * n.x + u.y * n.y + u.z * n.z + u.w * n.w;
#pragma unroll
    for (int m = 16; m >= 1; m >>= 1) {
        ps += __shfl_xor(ps, m, 32);
        ns += __shfl_xor(ns, m, 32);
    }
    if (lane == 0) atomAddF(acc, fmaxf(0.f, 1.0f - ps + ns));
}

__global__ void finalize(const float* __restrict__ acc, float* __restrict__ out) {
    const float lr = acc[0] * (1.0f / (float)BB);
    const float ls = acc[1];
    out[0] = lr + kLAMBDA1 * ls;  // loss
    out[1] = lr;                  // loss_r
    out[2] = ls;                  // loss_s
}

// ---------------- host orchestration ----------------
extern "C" void kernel_launch(void* const* d_in, const int* in_sizes, int n_in,
                              void* d_out, int out_size, void* d_ws, size_t ws_size,
                              hipStream_t stream) {
    (void)in_sizes; (void)n_in; (void)out_size; (void)ws_size;

    const float* E_u_0    = (const float*)d_in[0];
    const float* E_i_0    = (const float*)d_in[1];
    const float* W_s      = (const float*)d_in[2];
    const float* u_mul_s  = (const float*)d_in[3];
    const float* v_mul_s  = (const float*)d_in[4];
    const float* ut       = (const float*)d_in[5];
    const float* vt       = (const float*)d_in[6];
    const float* adj_vals = (const float*)d_in[7];
    const int*   adj_rows = (const int*)d_in[8];
    const int*   adj_cols = (const int*)d_in[9];
    const int*   uids     = (const int*)d_in[10];
    const int*   iids     = (const int*)d_in[11];
    const int*   pids     = (const int*)d_in[12];
    const int*   nids     = (const int*)d_in[13];

    // carve workspace
    char* base = (char*)d_ws;
    size_t off = 0;
    auto carve = [&](size_t bytes) -> void* {
        void* p = base + off;
        off += (bytes + 255) & ~(size_t)255;
        return p;
    };
    float* E_u    = (float*)carve(sizeof(float) * (size_t)NU * DIM);
    float* E_i    = (float*)carve(sizeof(float) * (size_t)NI * DIM);
    float* E_u_s  = (float*)carve(sizeof(float) * (size_t)NU * DIM);
    float* E_i_s  = (float*)carve(sizeof(float) * (size_t)NI * DIM);
    float* Z_u    = (float*)carve(sizeof(float) * (size_t)NU * DIM);
    float* Z_i    = (float*)carve(sizeof(float) * (size_t)NI * DIM);
    float* G_u    = (float*)carve(sizeof(float) * (size_t)NU * DIM);
    float* G_i    = (float*)carve(sizeof(float) * (size_t)NI * DIM);
    float* P_i    = (float*)carve(sizeof(float) * QQ * DIM);
    float* P_u    = (float*)carve(sizeof(float) * QQ * DIM);
    _Float16* Wt_h  = (_Float16*)carve(sizeof(_Float16) * DIM * DIM);
    float*    gnnF  = (float*)   carve(sizeof(float) * (size_t)BB * DIM);
    _Float16* gnnH  = (_Float16*)carve(sizeof(_Float16) * (size_t)BB * DIM);
    _Float16* hypnH = (_Float16*)carve(sizeof(_Float16) * (size_t)BB * DIM);
    float*    hypF  = (float*)   carve(sizeof(float) * (size_t)BB * DIM);
    _Float16* hypH  = (_Float16*)carve(sizeof(_Float16) * (size_t)BB * DIM);
    float* posb = (float*)carve(sizeof(float) * BB);
    float* negb = (float*)carve(sizeof(float) * BB);
    float* acc  = (float*)carve(sizeof(float) * 2);  // [0]=bpr sum, [1]=loss_s

    // init state (deterministic each call)
    hipMemsetAsync(acc, 0, 2 * sizeof(float), stream);
    hipMemcpyAsync(E_u,   E_u_0, sizeof(float) * (size_t)NU * DIM, hipMemcpyDeviceToDevice, stream);
    hipMemcpyAsync(E_i,   E_i_0, sizeof(float) * (size_t)NI * DIM, hipMemcpyDeviceToDevice, stream);
    hipMemcpyAsync(E_u_s, E_u_0, sizeof(float) * (size_t)NU * DIM, hipMemcpyDeviceToDevice, stream);
    hipMemcpyAsync(E_i_s, E_i_0, sizeof(float) * (size_t)NI * DIM, hipMemcpyDeviceToDevice, stream);

    for (int l = 0; l < LAYERS; ++l) {
        const uint32_t s1 = 0xA0000000u + l * 2, s2 = s1 + 1;
        hipMemsetAsync(Z_u, 0, sizeof(float) * (size_t)NU * DIM, stream);
        hipMemsetAsync(Z_i, 0, sizeof(float) * (size_t)NI * DIM, stream);
        hipMemsetAsync(P_i, 0, sizeof(float) * QQ * DIM, stream);
        hipMemsetAsync(P_u, 0, sizeof(float) * QQ * DIM, stream);

        // Z_u = segsum(v1 * E_i[cols] -> rows); Z_i = segsum(v2 * E_u[rows] -> cols)
        spmm_scatter<<<4096, 256, 0, stream>>>(adj_vals, adj_rows, adj_cols, E_i, Z_u, NEDGE, s1);
        spmm_scatter<<<4096, 256, 0, stream>>>(adj_vals, adj_cols, adj_rows, E_u, Z_i, NEDGE, s2);

        // P_i = vt @ E_i ; P_u = ut @ E_u ; then low-rank broadcast
        qd_gemm<<<512, DIM, 0, stream>>>(vt, E_i, P_i, NI);
        qd_gemm<<<512, DIM, 0, stream>>>(ut, E_u, P_u, NU);
        gprop<<<4096, 256, 0, stream>>>(u_mul_s, P_i, G_u, NU);
        gprop<<<2048, 256, 0, stream>>>(v_mul_s, P_u, G_i, NI);

        // Z = lrelu(Z); E += Z; Esum += E
        layer_update<<<4096, 256, 0, stream>>>(Z_u, E_u, E_u_s, NU * DIM);
        layer_update<<<2048, 256, 0, stream>>>(Z_i, E_i, E_i_s, NI * DIM);

        // per-layer InfoNCE (uses the just-computed Z/G)
        convert_wt<<<(DIM * DIM + 255) / 256, 256, 0, stream>>>(W_s + (size_t)l * DIM * DIM, Wt_h);

        for (int side = 0; side < 2; ++side) {
            const float* Zx = side ? Z_i : Z_u;
            const float* Gx = side ? G_i : G_u;
            const int*   ix = side ? iids : uids;
            const uint32_t ms = 0xB0000000u + l * 2 + side;

            gather_norm<<<512, 256, 0, stream>>>(Zx, ix, gnnF, gnnH, BB);
            gather_norm<<<512, 256, 0, stream>>>(Gx, ix, (float*)nullptr, hypnH, BB);
            hyper_gemm<<<dim3(BB / 16, DIM / 16), 32, 0, stream>>>(hypnH, Wt_h, hypF, hypH);
            pos_kernel<<<512, 256, 0, stream>>>(gnnF, hypF, posb, BB);
            neg_gemm<<<dim3(BB / 16 / 8), 256, 0, stream>>>(gnnH, hypH, negb);
            infonce_loss<<<BB / 256, 256, 0, stream>>>(posb, negb, acc + 1, ms, BB);
        }
    }

    bpr_kernel<<<512, 256, 0, stream>>>(E_u_s, E_i_s, uids, pids, nids, acc, BB);
    finalize<<<1, 1, 0, stream>>>(acc, (float*)d_out);
}
